// Model_23974507446686
// MI455X (gfx1250) — matrix operand order repaired and checked by reading, not compiled, not run on hardware
//
#include <hip/hip_runtime.h>
#include <type_traits>

typedef __attribute__((ext_vector_type(16))) _Float16 v16h;
typedef __attribute__((ext_vector_type(8)))  _Float16 v8h;
typedef __attribute__((ext_vector_type(8)))  float    v8f;

#define FLAG_RELU  1
#define FLAG_RESID 2
#define FLAG_CT    4
#define FLAG_BHEAD 8

__device__ __forceinline__ v8f wmma_f16(v16h a, v16h b, v8f c) {
  // D = A(16x32 f16) * B(32x16 f16) + C(16x16 f32)
  return __builtin_amdgcn_wmma_f32_16x16x32_f16(false, a, false, b, (short)0, c,
                                                false, false);
}

// A-fragment load pattern: halves 0..7 at p[0..7], halves 8..15 at p[16..23]
// (caller folds the per-lane row*ld + 8*(lane>=16) offset into p).
__device__ __forceinline__ v16h ldfrag_gap(const _Float16* p) {
  v16h a;
  ((uint4*)&a)[0] = *(const uint4*)(p);
  ((uint4*)&a)[1] = *(const uint4*)(p + 16);
  return a;
}

// B-fragment load pattern: 16 contiguous halves
// (caller folds the per-lane col*ld + 16*(lane>=16) offset into p).
__device__ __forceinline__ v16h ldfrag_cont(const _Float16* p) {
  v16h b;
  ((uint4*)&b)[0] = *(const uint4*)(p);
  ((uint4*)&b)[1] = *(const uint4*)(p + 8);
  return b;
}

__device__ __forceinline__ v16h frag_a(const _Float16* base, int row0, int ld,
                                       int k0, int lane) {
  return ldfrag_gap(base + (size_t)(row0 + (lane & 15)) * ld + k0 +
                    ((lane >> 4) << 3));
}

__device__ __forceinline__ v16h frag_bt(const _Float16* base, int n0, int ld,
                                        int k0, int lane) {
  return ldfrag_gap(base + (size_t)(n0 + (lane & 15)) * ld + k0 +
                    ((lane >> 4) << 3));
}

// ---------------------------------------------------------------------------
// Generic GEMM: C = act(A * B + bias) (+ resid), f16 WMMA core, f32 accum.
// Block tile 128x128, K-tile 32, 8 waves (wave32), wave tile 32x64 (2x4 WMMA).
// Software-pipelined: K-tile k+1 is fetched into registers while tile k is
// consumed from LDS by the WMMA stream.
// FLAG_BHEAD: B is stored head-blocked [n/64][k][64] with head stride ldb.
// FLAG_CT:    store C transposed (C[col*ldc + row]).
// ---------------------------------------------------------------------------
template <typename AT, typename OT>
__global__ __launch_bounds__(256) void gemm_wmma(
    const AT* __restrict__ A, int lda, const float* __restrict__ Bw, int ldb,
    OT* __restrict__ C, int ldc, const float* __restrict__ bias,
    const float* __restrict__ resid, int ldr, int M, int N, int K, int flags) {
  __shared__ _Float16 As[128][40];  // padded rows vs bank conflicts
  __shared__ _Float16 Bs[128][40];  // stored transposed [n][k]

  const int tid = threadIdx.x;
  const int lane = tid & 31, wave = tid >> 5;
  const int wm = wave >> 1, wn = wave & 1;
  const int m0 = blockIdx.y * 128;
  const int n0 = blockIdx.x * 128;

  v8f zero = {};
  v8f acc[2][4];
#pragma unroll
  for (int i = 0; i < 2; ++i)
#pragma unroll
    for (int j = 0; j < 4; ++j) acc[i][j] = zero;

  float4 ra[4];  // global-fetch staging registers (AT == float)
  uint4 rah[2];  //                                (AT == f16)
  float4 rb[4];

  auto gload = [&](int kk0) {
    if constexpr (std::is_same<AT, float>::value) {
#pragma unroll
      for (int i = 0; i < 4; ++i) {
        int fid = tid + i * 256;  // 1024 float4 chunks of 128x32
        int row = fid >> 3, c4 = (fid & 7) << 2;
        ra[i] = *(const float4*)(A + (size_t)(m0 + row) * lda + kk0 + c4);
      }
    } else {
#pragma unroll
      for (int i = 0; i < 2; ++i) {
        int cid = tid + i * 256;  // 512 16-byte chunks
        int row = cid >> 2, c8 = (cid & 3) << 3;
        rah[i] = *(const uint4*)(A + (size_t)(m0 + row) * lda + kk0 + c8);
      }
    }
#pragma unroll
    for (int i = 0; i < 4; ++i) {
      int fid = tid + i * 256;  // 1024 float4 chunks of 32x128
      int kk = fid >> 5, n4 = (fid & 31) << 2;
      const float* bp;
      if (flags & FLAG_BHEAD) {
        int n = n0 + n4;  // head-blocked: [n>>6][k][n&63]
        bp = Bw + (size_t)(n >> 6) * ldb + (size_t)(kk0 + kk) * 64 + (n & 63);
      } else {
        bp = Bw + (size_t)(kk0 + kk) * ldb + n0 + n4;
      }
      rb[i] = *(const float4*)bp;
    }
  };

  auto lcommit = [&]() {
    if constexpr (std::is_same<AT, float>::value) {
#pragma unroll
      for (int i = 0; i < 4; ++i) {
        int fid = tid + i * 256;
        int row = fid >> 3, c4 = (fid & 7) << 2;
        As[row][c4 + 0] = (_Float16)ra[i].x;
        As[row][c4 + 1] = (_Float16)ra[i].y;
        As[row][c4 + 2] = (_Float16)ra[i].z;
        As[row][c4 + 3] = (_Float16)ra[i].w;
      }
    } else {
#pragma unroll
      for (int i = 0; i < 2; ++i) {
        int cid = tid + i * 256;
        int row = cid >> 2, c8 = (cid & 3) << 3;
        *(uint4*)&As[row][c8] = rah[i];
      }
    }
#pragma unroll
    for (int i = 0; i < 4; ++i) {
      int fid = tid + i * 256;
      int kk = fid >> 5, n4 = (fid & 31) << 2;
      Bs[n4 + 0][kk] = (_Float16)rb[i].x;
      Bs[n4 + 1][kk] = (_Float16)rb[i].y;
      Bs[n4 + 2][kk] = (_Float16)rb[i].z;
      Bs[n4 + 3][kk] = (_Float16)rb[i].w;
    }
  };

  gload(0);
  lcommit();
  __syncthreads();

  for (int k0 = 0; k0 < K; k0 += 32) {
    const bool more = (k0 + 32) < K;
    if (more) {
      gload(k0 + 32);  // overlap next tile's global fetch with WMMA below
      if (k0 + 64 < K)  // speculative prefetch two tiles ahead
        __builtin_prefetch(A + (size_t)(m0 + (tid >> 1)) * lda + k0 + 64, 0, 1);
    }
    v16h a0 = frag_a(&As[0][0], wm * 32 + 0, 40, 0, lane);
    v16h a1 = frag_a(&As[0][0], wm * 32 + 16, 40, 0, lane);
#pragma unroll
    for (int j = 0; j < 4; ++j) {
      v16h bj = frag_bt(&Bs[0][0], wn * 64 + j * 16, 40, 0, lane);
      acc[0][j] = wmma_f16(a0, bj, acc[0][j]);
      acc[1][j] = wmma_f16(a1, bj, acc[1][j]);
    }
    __syncthreads();
    if (more) {
      lcommit();
      __syncthreads();
    }
  }

  // ---- epilogue: bias / relu / residual, normal or transposed store ----
  const int half = lane >> 4, nc = lane & 15;
#pragma unroll
  for (int ti = 0; ti < 2; ++ti)
#pragma unroll
    for (int tj = 0; tj < 4; ++tj) {
      int col = n0 + wn * 64 + tj * 16 + nc;
#pragma unroll
      for (int r = 0; r < 8; ++r) {
        int row = m0 + wm * 32 + ti * 16 + r + half * 8;
        float v = acc[ti][tj][r];
        if (bias) v += bias[col];
        if (flags & FLAG_RELU) v = fmaxf(v, 0.f);
        if (flags & FLAG_RESID) v += resid[(size_t)row * ldr + col];
        if (flags & FLAG_CT)
          C[(size_t)col * ldc + row] = (OT)v;
        else
          C[(size_t)row * ldc + col] = (OT)v;
      }
    }
}

// ---------------------------------------------------------------------------
// Flash attention, causal, computed TRANSPOSED so softmax is in-register:
//   S^T = K * Q^T   (lane = one query column, regs = 8 keys)
//   O^T = V^T * P   (accumulated transposed, stored contiguously)
// q,k: f16 [B*T][H*64] ; vt: f16 [H*64][B*T] ; o: f32 [B*T][D]
// One wave per 16 query rows; key blocks of 32; 2 shuffles per block.
// All fragment pointers are per-lane bases bumped by constants per block so
// offsets fold into the instruction's 24-bit ioffset.
// ---------------------------------------------------------------------------
__global__ __launch_bounds__(128) void attn_flash(
    const _Float16* __restrict__ q, const _Float16* __restrict__ k,
    const _Float16* __restrict__ vt, float* __restrict__ o, int Bn, int T,
    int H, int D) {
  __shared__ _Float16 Pb[4][16][40];  // per-wave P[q][key] re-layout buffer
  const int tid = threadIdx.x, lane = tid & 31, wave = tid >> 5;
  const int bh = blockIdx.y;
  const int b = bh % Bn, h = bh / Bn;
  const int M = Bn * T;
  const int t0 = (blockIdx.x * 4 + wave) * 16;
  const int half = lane >> 4, nc = lane & 15;

  const _Float16* qb = q + (size_t)b * T * D + h * 64;
  // per-lane fragment base pointers (A and B operands both use the gap pattern: +8*half)
  const _Float16* kl = k + (size_t)b * T * D + h * 64 +
                       (size_t)(lane & 15) * D + (half << 3);
  const _Float16* vl = vt + (size_t)h * 64 * M + (size_t)b * T +
                       (size_t)(lane & 15) * M + (half << 3);
  const _Float16* pl = &Pb[wave][0][0] + nc * 40 + (half << 3);
  _Float16* pw = &Pb[wave][0][0] + nc * 40 + (half << 3);

  // Q^T as the B operand (rows = queries, contraction e contiguous)
  v16h qfb0 = frag_bt(qb, t0, D, 0, lane);
  v16h qfb1 = frag_bt(qb, t0, D, 32, lane);

  v8f zero = {};
  v8f accO[4];  // O^T tiles: rows e = j*16 + r + 8*half, col = query nc
#pragma unroll
  for (int j = 0; j < 4; ++j) accO[j] = zero;
  float mrun = -__builtin_inff(), lrun = 0.f;  // per-query scalars

  const float scale = 0.125f;  // 1/sqrt(64)
  for (int tk0 = 0; tk0 < t0 + 16; tk0 += 32, kl += 32 * D, vl += 32) {
    // hoist all four K fragments, then chain the WMMAs
    v16h k00 = ldfrag_gap(kl);
    v16h k01 = ldfrag_gap(kl + 32);
    v16h k10 = ldfrag_gap(kl + 16 * D);
    v16h k11 = ldfrag_gap(kl + 16 * D + 32);
    v8f st0 = zero, st1 = zero;
    st0 = wmma_f16(k00, qfb0, st0);
    st0 = wmma_f16(k01, qfb1, st0);
    st1 = wmma_f16(k10, qfb0, st1);
    st1 = wmma_f16(k11, qfb1, st1);
    // issue V fragment loads now; latency hides under the softmax VALU work
    v16h v0 = ldfrag_gap(vl);
    v16h v1 = ldfrag_gap(vl + 16 * M);
    v16h v2 = ldfrag_gap(vl + 32 * M);
    v16h v3 = ldfrag_gap(vl + 48 * M);

    const bool need_mask = (tk0 + 31 > t0);
    const int mg = t0 + nc;  // this lane's query index
    float mx = -__builtin_inff();
#pragma unroll
    for (int r = 0; r < 8; ++r) {
      float a = st0[r] * scale, c = st1[r] * scale;
      if (need_mask) {
        if (tk0 + r + half * 8 > mg) a = -__builtin_inff();
        if (tk0 + 16 + r + half * 8 > mg) c = -__builtin_inff();
      }
      st0[r] = a;
      st1[r] = c;
      mx = fmaxf(mx, fmaxf(a, c));
    }
    mx = fmaxf(mx, __shfl_xor(mx, 16, 32));  // combine the two lane halves

    float mn = fmaxf(mrun, mx);
    float alpha = __expf(mrun - mn);
    float ls = 0.f;
    v8h p0, p1;  // 8 consecutive keys per lane -> packed 16B LDS stores
#pragma unroll
    for (int r = 0; r < 8; ++r) {
      float pa = __expf(st0[r] - mn);
      float pb = __expf(st1[r] - mn);
      ls += pa + pb;
      p0[r] = (_Float16)pa;
      p1[r] = (_Float16)pb;
    }
    ls += __shfl_xor(ls, 16, 32);
    lrun = lrun * alpha + ls;
    mrun = mn;
#pragma unroll
    for (int j = 0; j < 4; ++j) accO[j] *= alpha;  // scalar rescale per lane

    // P[q][key]: row = query nc, keys contiguous -> two ds_store_b128
    *(v8h*)(pw) = p0;
    *(v8h*)(pw + 16) = p1;
    asm volatile("s_wait_dscnt 0" ::: "memory");  // same-wave LDS drain
    v16h pfb = ldfrag_gap(pl);
    accO[0] = wmma_f16(v0, pfb, accO[0]);  // O^T += V^T * P
    accO[1] = wmma_f16(v1, pfb, accO[1]);
    accO[2] = wmma_f16(v2, pfb, accO[2]);
    accO[3] = wmma_f16(v3, pfb, accO[3]);
    asm volatile("" ::: "memory");
  }

  // O^T: each lane owns query row (t0+nc), 8 consecutive e per tile
  const float inv = 1.f / lrun;
  float* op = o + (size_t)(b * T + t0 + nc) * D + h * 64 + half * 8;
#pragma unroll
  for (int j = 0; j < 4; ++j) {
    float4 lo, hi;
    lo.x = accO[j][0] * inv; lo.y = accO[j][1] * inv;
    lo.z = accO[j][2] * inv; lo.w = accO[j][3] * inv;
    hi.x = accO[j][4] * inv; hi.y = accO[j][5] * inv;
    hi.z = accO[j][6] * inv; hi.w = accO[j][7] * inv;
    *(float4*)(op + j * 16) = lo;
    *(float4*)(op + j * 16 + 4) = hi;
  }
}

// ---------------------------------------------------------------------------
// LayerNorm: one 256-thread block per row, D=1024 (one float4 per thread).
// ---------------------------------------------------------------------------
__global__ __launch_bounds__(256) void layernorm(const float* __restrict__ x,
                                                 const float* __restrict__ g,
                                                 const float* __restrict__ bt,
                                                 float* __restrict__ y, int D) {
  __shared__ float red[16];
  const int row = blockIdx.x, tid = threadIdx.x;
  const float* xr = x + (size_t)row * D;
  float4 f = *(const float4*)(xr + tid * 4);
  float s = f.x + f.y + f.z + f.w;
  float ss = f.x * f.x + f.y * f.y + f.z * f.z + f.w * f.w;
#pragma unroll
  for (int m = 1; m < 32; m <<= 1) {
    s += __shfl_xor(s, m, 32);
    ss += __shfl_xor(ss, m, 32);
  }
  if ((tid & 31) == 0) {
    red[tid >> 5] = s;
    red[8 + (tid >> 5)] = ss;
  }
  __syncthreads();
  float ts = 0.f, tss = 0.f;
#pragma unroll
  for (int i = 0; i < 8; ++i) {
    ts += red[i];
    tss += red[8 + i];
  }
  float mean = ts / D;
  float inv = rsqrtf(tss / D - mean * mean + 1e-5f);
  float4 gg = *(const float4*)(g + tid * 4);
  float4 bb = *(const float4*)(bt + tid * 4);
  float4 out;
  out.x = (f.x - mean) * inv * gg.x + bb.x;
  out.y = (f.y - mean) * inv * gg.y + bb.y;
  out.z = (f.z - mean) * inv * gg.z + bb.z;
  out.w = (f.w - mean) * inv * gg.w + bb.w;
  *(float4*)(y + (size_t)row * D + tid * 4) = out;
}

// h[row] = tok_embd[id] + pos_embd[id]  (reference's quirk: pos indexed by id)
__global__ __launch_bounds__(256) void embed(const int* __restrict__ x,
                                             const float* __restrict__ tok,
                                             const float* __restrict__ pos,
                                             float* __restrict__ h, int D) {
  int row = blockIdx.x;
  int id = x[row];
  const float4* t = (const float4*)(tok + (size_t)id * D);
  const float4* p = (const float4*)(pos + (size_t)id * D);
  float4* hd = (float4*)(h + (size_t)row * D);
  int i = threadIdx.x;  // D/4 == 256
  float4 a = t[i], b = p[i];
  float4 o;
  o.x = a.x + b.x; o.y = a.y + b.y; o.z = a.z + b.z; o.w = a.w + b.w;
  hd[i] = o;
}

extern "C" void kernel_launch(void* const* d_in, const int* in_sizes, int n_in,
                              void* d_out, int out_size, void* d_ws,
                              size_t ws_size, hipStream_t stream) {
  (void)in_sizes; (void)n_in; (void)out_size; (void)ws_size;
  constexpr int Bn = 4, T = 1024, D = 1024, H = 16, L = 8, HS = 64, V = 8192;
  constexpr int M = Bn * T, F = 4 * D;

  const int* x = (const int*)d_in[0];
  const float* tok = (const float*)d_in[1];
  const float* pos = (const float*)d_in[2];
  const float* wq = (const float*)d_in[3];
  const float* wk = (const float*)d_in[4];
  const float* wv = (const float*)d_in[5];
  const float* w_o = (const float*)d_in[6];
  const float* b_o = (const float*)d_in[7];
  const float* ln1g = (const float*)d_in[8];
  const float* ln1b = (const float*)d_in[9];
  const float* ln2g = (const float*)d_in[10];
  const float* ln2b = (const float*)d_in[11];
  const float* w1 = (const float*)d_in[12];
  const float* b1 = (const float*)d_in[13];
  const float* w2 = (const float*)d_in[14];
  const float* b2 = (const float*)d_in[15];
  const float* lnfg = (const float*)d_in[16];
  const float* lnfb = (const float*)d_in[17];
  const float* w_lm = (const float*)d_in[18];
  const float* b_lm = (const float*)d_in[19];

  char* ws = (char*)d_ws;
  float* h = (float*)ws;      ws += (size_t)M * D * 4;
  float* hn = (float*)ws;     ws += (size_t)M * D * 4;
  float* ob = (float*)ws;     ws += (size_t)M * D * 4;
  _Float16* qb = (_Float16*)ws;  ws += (size_t)M * D * 2;   // [M][H*64]
  _Float16* kb = (_Float16*)ws;  ws += (size_t)M * D * 2;   // [M][H*64]
  _Float16* vtb = (_Float16*)ws; ws += (size_t)M * D * 2;   // [H*64][M]
  _Float16* ff = (_Float16*)ws;  ws += (size_t)M * F * 2;

  embed<<<M, 256, 0, stream>>>(x, tok, pos, h, D);

  for (int l = 0; l < L; ++l) {
    layernorm<<<M, 256, 0, stream>>>(h, ln1g + (size_t)l * D,
                                     ln1b + (size_t)l * D, hn, D);
    const float* wql = wq + (size_t)l * H * D * HS;
    const float* wkl = wk + (size_t)l * H * D * HS;
    const float* wvl = wv + (size_t)l * H * D * HS;
    // Q,K: f16 [M][H*64]; head-blocked weight, ldb = head stride = D*HS
    gemm_wmma<float, _Float16><<<dim3(D / 128, M / 128), 256, 0, stream>>>(
        hn, D, wql, D * HS, qb, D, nullptr, nullptr, 0, M, D, D, FLAG_BHEAD);
    gemm_wmma<float, _Float16><<<dim3(D / 128, M / 128), 256, 0, stream>>>(
        hn, D, wkl, D * HS, kb, D, nullptr, nullptr, 0, M, D, D, FLAG_BHEAD);
    // V stored transposed: f16 [H*64][M]
    gemm_wmma<float, _Float16><<<dim3(D / 128, M / 128), 256, 0, stream>>>(
        hn, D, wvl, D * HS, vtb, M, nullptr, nullptr, 0, M, D, D,
        FLAG_BHEAD | FLAG_CT);
    attn_flash<<<dim3(T / 64, Bn * H), 128, 0, stream>>>(qb, kb, vtb, ob, Bn,
                                                         T, H, D);
    // h = hn + o @ w_o + b_o   (residual from hn, per reference quirk)
    gemm_wmma<float, float><<<dim3(D / 128, M / 128), 256, 0, stream>>>(
        ob, D, w_o + (size_t)l * D * D, D, h, D, b_o + (size_t)l * D, hn, D,
        M, D, D, FLAG_RESID);
    layernorm<<<M, 256, 0, stream>>>(h, ln2g + (size_t)l * D,
                                     ln2b + (size_t)l * D, hn, D);
    // ff = relu(hn @ w1 + b1)  (f16)
    gemm_wmma<float, _Float16><<<dim3(F / 128, M / 128), 256, 0, stream>>>(
        hn, D, w1 + (size_t)l * D * F, F, ff, F, b1 + (size_t)l * F, nullptr,
        0, M, F, D, FLAG_RELU);
    // h = hn + ff @ w2 + b2
    gemm_wmma<_Float16, float><<<dim3(D / 128, M / 128), 256, 0, stream>>>(
        ff, F, w2 + (size_t)l * F * D, D, h, D, b2 + (size_t)l * D, hn, D,
        M, D, F, FLAG_RESID);
  }

  layernorm<<<M, 256, 0, stream>>>(h, lnfg, lnfb, hn, D);
  gemm_wmma<float, float><<<dim3(V / 128, M / 128), 256, 0, stream>>>(
      hn, D, w_lm, V, (float*)d_out, V, b_lm, nullptr, 0, M, V, D, 0);
}
